// MAE_47983374631200
// MI455X (gfx1250) — compile-verified, matrix operand order checked by reading, not run on hardware
//
#include <hip/hip_runtime.h>
#include <hip/hip_bf16.h>
#include <math.h>

// ---------------------------------------------------------------------------
// Model dimensions (compile-time constants from the reference)
// ---------------------------------------------------------------------------
static constexpr int Bc   = 32;
static constexpr int Nn   = 256;
static constexpr int PP   = 256;    // P*P
static constexpr int KEEPc= 64;
static constexpr int MSKc = 192;
static constexpr int DEc  = 768;
static constexpr int DDc  = 512;
static constexpr int SE   = 65;     // encoder seq
static constexpr int SD   = 257;    // decoder seq
static constexpr int LE   = 12;
static constexpr int LD   = 4;
static constexpr int HE   = 12;     // heads
static constexpr int HD   = 8;
static constexpr int HDIM = 64;     // head dim (both stacks)

typedef __attribute__((ext_vector_type(8)))  float  v8f;
typedef __attribute__((ext_vector_type(16))) __bf16 v16bf;

// ---------------------------------------------------------------------------
// helpers
// ---------------------------------------------------------------------------
__device__ __forceinline__ unsigned f2bf(float f) {
    union { float f; unsigned u; } c; c.f = f;
    unsigned u = c.u;
    unsigned r = u + 0x7FFFu + ((u >> 16) & 1u);   // round-to-nearest-even
    return r >> 16;
}
__device__ __forceinline__ unsigned pack2(float lo, float hi) {
    return f2bf(lo) | (f2bf(hi) << 16);
}

// fp32 -> bf16 weight pre-conversion (once per call)
__global__ __launch_bounds__(256) void k_to_bf16(const float* __restrict__ src,
                                                 unsigned short* __restrict__ dst, int n) {
    int idx = blockIdx.x * 256 + threadIdx.x;
    if (idx < n) dst[idx] = (unsigned short)f2bf(src[idx]);
}

// ---------------------------------------------------------------------------
// Global mean / var of x  (two stage)
// ---------------------------------------------------------------------------
__global__ __launch_bounds__(256) void k_reduce_partial(const float* __restrict__ x,
                                                        float* __restrict__ partial, int n) {
    __shared__ float s1[256], s2[256];
    int tid = threadIdx.x;
    float a = 0.f, b = 0.f;
    for (int i = blockIdx.x * 256 + tid; i < n; i += 256 * 256) {
        float v = x[i]; a += v; b += v * v;
    }
    s1[tid] = a; s2[tid] = b; __syncthreads();
    for (int off = 128; off > 0; off >>= 1) {
        if (tid < off) { s1[tid] += s1[tid + off]; s2[tid] += s2[tid + off]; }
        __syncthreads();
    }
    if (tid == 0) { partial[blockIdx.x] = s1[0]; partial[256 + blockIdx.x] = s2[0]; }
}

__global__ __launch_bounds__(256) void k_reduce_final(const float* __restrict__ partial,
                                                      float* __restrict__ stats, int n) {
    __shared__ float s1[256], s2[256];
    int tid = threadIdx.x;
    s1[tid] = partial[tid]; s2[tid] = partial[256 + tid]; __syncthreads();
    for (int off = 128; off > 0; off >>= 1) {
        if (tid < off) { s1[tid] += s1[tid + off]; s2[tid] += s2[tid + off]; }
        __syncthreads();
    }
    if (tid == 0) {
        float mean = s1[0] / (float)n;
        float var  = s2[0] / (float)n - mean * mean;
        stats[0] = mean;
        stats[1] = rsqrtf(var + 1e-5f) * 0.5f;   // fold the 0.5 scale in
    }
}

// patches[b,n,p] = xn rearranged  (n = hb*16+wb, p = ph*16+pw)
__global__ __launch_bounds__(256) void k_patchify(const float* __restrict__ x,
                                                  const float* __restrict__ stats,
                                                  float* __restrict__ patches) {
    int idx = blockIdx.x * 256 + threadIdx.x;
    if (idx >= Bc * Nn * PP) return;
    int p  = idx % PP;
    int n  = (idx / PP) % Nn;
    int b  = idx / (PP * Nn);
    int ph = p >> 4, pw = p & 15;
    int hb = n >> 4, wb = n & 15;
    float v = x[((size_t)b * 256 + hb * 16 + ph) * 256 + wb * 16 + pw];
    patches[idx] = (v - stats[0]) * stats[1];
}

// ---------------------------------------------------------------------------
// 2D sincos positional embedding: pe[0]=0, pe[1+t] = [sin_i|cos_i|sin_j|cos_j]
// ---------------------------------------------------------------------------
__global__ __launch_bounds__(256) void k_pos_emb(float* __restrict__ pe, int rows, int dim) {
    int idx = blockIdx.x * 256 + threadIdx.x;
    if (idx >= rows * dim) return;
    int row = idx / dim, col = idx % dim;
    if (row == 0) { pe[idx] = 0.f; return; }
    int t = row - 1;
    int i = t >> 4, j = t & 15;
    int quarter = dim >> 2;
    int which = col / quarter;
    int k = col % quarter;
    float om  = powf(10000.f, -(float)k / (float)quarter);
    float pos = (which < 2) ? (float)i : (float)j;
    float ang = pos * om;
    pe[idx] = ((which & 1) == 0) ? sinf(ang) : cosf(ang);
}

// ---------------------------------------------------------------------------
// bf16 WMMA GEMM:  C[M,N] = A[M,K](fp32) * Wb[N,K](bf16)^T (+bias)(+res)(gelu)
// tile: 128x128x32, 256 threads = 8 waves (4 along M, 2 along N),
// each wave owns 32x64 = 2x4 fragments of v_wmma_f32_16x16x32_bf16.
// A is converted fp32->bf16 in registers while staging to LDS.
// W is pre-converted bf16 and staged with global_load_async_to_lds_b128
// (ASYNCcnt path), synced via s_wait_asynccnt + workgroup barrier.
// LDS rows are 40 ushorts (80B): 16B-aligned for b128 ops, 20-bank stride.
// ---------------------------------------------------------------------------
#define GBM 128
#define GBN 128
#define GBK 32
#define LDSW 40

__global__ __launch_bounds__(256) void k_gemm_wmma(const float* __restrict__ A,
                                                   const unsigned short* __restrict__ Wb,
                                                   float* __restrict__ C,
                                                   const float* __restrict__ bias,
                                                   const float* __restrict__ residual,
                                                   int M, int N, int K, int gelu_flag) {
    __shared__ unsigned short As[GBM][LDSW];
    __shared__ unsigned short Bs[GBN][LDSW];

    int tid  = threadIdx.x;
    int wave = tid >> 5;
    int lane = tid & 31;
    int wm   = wave & 3;          // wave position along M (0..3) -> 32 rows each
    int wn   = wave >> 2;         // along N (0..1)              -> 64 cols each
    int m0   = blockIdx.y * GBM;
    int n0   = blockIdx.x * GBN;

    // A staging: 2 threads per row, 16 floats each
    int sr = tid >> 1;
    int sc = (tid & 1) * 16;
    // B async staging: 2 threads per row, 2x16B chunks each
    int br = tid >> 1;
    int bc = (tid & 1) * 2;      // 16B-chunk index (0 or 2)
    unsigned bsBase = (unsigned)(uintptr_t)&Bs[0][0];

    v8f acc[2][4] = {};

    for (int k0 = 0; k0 < K; k0 += GBK) {
        // ---- async-copy B tile (bf16 weights) straight into LDS ----
        {
            unsigned lds0 = bsBase + (unsigned)(br * (LDSW * 2) + bc * 16);
            const unsigned short* g = Wb + (size_t)(n0 + br) * K + k0 + bc * 8;
            asm volatile(
                "global_load_async_to_lds_b128 %0, %1, off\n\t"
                "global_load_async_to_lds_b128 %2, %1, off offset:16"
                :: "v"(lds0), "v"(g), "v"(lds0 + 16) : "memory");
        }
        // ---- stage A tile (fp32 -> packed bf16), one branch per thread ----
        {
            int gr = m0 + sr;
            unsigned u0, u1, u2, u3, u4, u5, u6, u7;
            if (gr < M) {
                const float4* s = (const float4*)(A + (size_t)gr * K + k0 + sc);
                float4 f0 = s[0], f1 = s[1], f2 = s[2], f3 = s[3];
                u0 = pack2(f0.x, f0.y); u1 = pack2(f0.z, f0.w);
                u2 = pack2(f1.x, f1.y); u3 = pack2(f1.z, f1.w);
                u4 = pack2(f2.x, f2.y); u5 = pack2(f2.z, f2.w);
                u6 = pack2(f3.x, f3.y); u7 = pack2(f3.z, f3.w);
            } else {
                u0 = u1 = u2 = u3 = u4 = u5 = u6 = u7 = 0u;
            }
            uint4* d = (uint4*)&As[sr][sc];
            d[0] = make_uint4(u0, u1, u2, u3);
            d[1] = make_uint4(u4, u5, u6, u7);
        }
        // wait for this wave's async copies; barrier covers all waves' work
        asm volatile("s_wait_asynccnt 0x0" ::: "memory");
        __syncthreads();

        // ---- build fragments: ISA 16-bit 16x32 layout (K half-ranges per lane group)
        int mr  = lane & 15;
        int kb0 = (lane < 16) ? 0 : 8;   // ushort index of first 8-elem K run
        int kb1 = kb0 + 16;

        v16bf afr[2], bfr[4];
#pragma unroll
        for (int mf = 0; mf < 2; ++mf) {
            const unsigned short* rp = &As[wm * 32 + mf * 16 + mr][0];
            uint4 lo = *(const uint4*)(rp + kb0);
            uint4 hi = *(const uint4*)(rp + kb1);
            __builtin_memcpy(&afr[mf], &lo, 16);
            __builtin_memcpy((char*)&afr[mf] + 16, &hi, 16);
        }
#pragma unroll
        for (int nf = 0; nf < 4; ++nf) {
            const unsigned short* rp = &Bs[wn * 64 + nf * 16 + mr][0];
            uint4 lo = *(const uint4*)(rp + kb0);
            uint4 hi = *(const uint4*)(rp + kb1);
            __builtin_memcpy(&bfr[nf], &lo, 16);
            __builtin_memcpy((char*)&bfr[nf] + 16, &hi, 16);
        }

#pragma unroll
        for (int mf = 0; mf < 2; ++mf)
#pragma unroll
            for (int nf = 0; nf < 4; ++nf)
                acc[mf][nf] = __builtin_amdgcn_wmma_f32_16x16x32_bf16(
                    false, afr[mf], false, bfr[nf], (short)0, acc[mf][nf], false, false);
        __syncthreads();
    }

    // epilogue: C/D layout -> VGPR r holds row (r + 8*(lane>=16)), col = lane%16
    int mhalf = (lane >= 16) ? 8 : 0;
    int ncol  = lane & 15;
#pragma unroll
    for (int mf = 0; mf < 2; ++mf)
#pragma unroll
        for (int nf = 0; nf < 4; ++nf)
#pragma unroll
            for (int r = 0; r < 8; ++r) {
                int gr = m0 + wm * 32 + mf * 16 + r + mhalf;
                int gc = n0 + wn * 64 + nf * 16 + ncol;
                if (gr < M) {
                    float v = acc[mf][nf][r];
                    if (bias)     v += bias[gc];
                    if (residual) v += residual[(size_t)gr * N + gc];
                    if (gelu_flag) v = 0.5f * v * (1.f + erff(v * 0.70710678118f));
                    C[(size_t)gr * N + gc] = v;
                }
            }
}

// ---------------------------------------------------------------------------
// LayerNorm: one block per row
// ---------------------------------------------------------------------------
__global__ __launch_bounds__(256) void k_layernorm(const float* __restrict__ x,
                                                   const float* __restrict__ g,
                                                   const float* __restrict__ b,
                                                   float* __restrict__ y, int D) {
    __shared__ float red[256];
    int row = blockIdx.x, tid = threadIdx.x;
    const float* xr = x + (size_t)row * D;
    float* yr = y + (size_t)row * D;
    float s = 0.f;
    for (int i = tid; i < D; i += 256) s += xr[i];
    red[tid] = s; __syncthreads();
    for (int off = 128; off > 0; off >>= 1) {
        if (tid < off) red[tid] += red[tid + off];
        __syncthreads();
    }
    float mean = red[0] / (float)D;
    __syncthreads();
    float sq = 0.f;
    for (int i = tid; i < D; i += 256) { float d = xr[i] - mean; sq += d * d; }
    red[tid] = sq; __syncthreads();
    for (int off = 128; off > 0; off >>= 1) {
        if (tid < off) red[tid] += red[tid + off];
        __syncthreads();
    }
    float rstd = rsqrtf(red[0] / (float)D + 1e-5f);
    for (int i = tid; i < D; i += 256) yr[i] = (xr[i] - mean) * rstd * g[i] + b[i];
}

// ---------------------------------------------------------------------------
// Attention: one thread per (b, head, query), online softmax, hd = 64
// qkv layout: [b, s, 3*D] with q|k|v blocks of D, head h at offset h*64
// ---------------------------------------------------------------------------
__global__ __launch_bounds__(128) void k_attention(const float* __restrict__ qkv,
                                                   float* __restrict__ o,
                                                   int S, int D, int heads) {
    int idx = blockIdx.x * 128 + threadIdx.x;
    int total = Bc * heads * S;
    if (idx >= total) return;
    int q   = idx % S;
    int tmp = idx / S;
    int h   = tmp % heads;
    int b   = tmp / heads;
    const float* base = qkv + (size_t)b * S * 3 * D;
    const float* qp   = base + (size_t)q * 3 * D + h * HDIM;

    float qreg[HDIM];
#pragma unroll
    for (int d = 0; d < HDIM; ++d) qreg[d] = qp[d];

    float m = -1e30f, l = 0.f;
    float oacc[HDIM];
#pragma unroll
    for (int d = 0; d < HDIM; ++d) oacc[d] = 0.f;

    for (int j = 0; j < S; ++j) {
        const float* kp = base + (size_t)j * 3 * D + D + h * HDIM;
        float s = 0.f;
#pragma unroll
        for (int d = 0; d < HDIM; ++d) s += qreg[d] * kp[d];
        s *= 0.125f;                            // 64^-0.5
        float nm   = fmaxf(m, s);
        float corr = __expf(m - nm);
        float p    = __expf(s - nm);
        l = l * corr + p;
        const float* vp = base + (size_t)j * 3 * D + 2 * D + h * HDIM;
#pragma unroll
        for (int d = 0; d < HDIM; ++d) oacc[d] = oacc[d] * corr + p * vp[d];
        m = nm;
    }
    float inv = 1.f / l;
    float* op = o + ((size_t)b * S + q) * D + h * HDIM;
#pragma unroll
    for (int d = 0; d < HDIM; ++d) op[d] = oacc[d] * inv;
}

// ---------------------------------------------------------------------------
// small elementwise kernels
// ---------------------------------------------------------------------------
__global__ __launch_bounds__(256) void k_add_pe(float* __restrict__ emb,
                                                const float* __restrict__ pe) {
    int idx = blockIdx.x * 256 + threadIdx.x;
    if (idx >= Bc * Nn * DEc) return;
    int c = idx % DEc;
    int n = (idx / DEc) % Nn;
    emb[idx] += pe[(size_t)(1 + n) * DEc + c];
}

__global__ __launch_bounds__(256) void k_build_encin(float* __restrict__ encin,
                                                     const float* __restrict__ emb,
                                                     const float* __restrict__ cls_enc,
                                                     const float* __restrict__ pe,
                                                     const int* __restrict__ uidx) {
    int idx = blockIdx.x * 256 + threadIdx.x;
    if (idx >= Bc * SE * DEc) return;
    int c = idx % DEc;
    int s = (idx / DEc) % SE;
    int b = idx / (DEc * SE);
    if (s == 0) encin[idx] = cls_enc[c] + pe[c];
    else {
        int n = uidx[b * KEEPc + s - 1];
        encin[idx] = emb[((size_t)b * Nn + n) * DEc + c];
    }
}

__global__ __launch_bounds__(256) void k_build_decin(float* __restrict__ decin,
                                                     const float* __restrict__ cls_dec,
                                                     const float* __restrict__ mask_emb,
                                                     const float* __restrict__ pe) {
    int idx = blockIdx.x * 256 + threadIdx.x;
    if (idx >= Bc * SD * DDc) return;
    int c = idx % DDc;
    int s = (idx / DDc) % SD;
    if (s == 0) decin[idx] = cls_dec[c] + pe[c];
    else        decin[idx] = mask_emb[c] + pe[(size_t)s * DDc + c];
}

__global__ __launch_bounds__(256) void k_scatter_unmasked(float* __restrict__ decin,
                                                          const float* __restrict__ proj,
                                                          const float* __restrict__ pe,
                                                          const int* __restrict__ uidx) {
    int idx = blockIdx.x * 256 + threadIdx.x;
    if (idx >= Bc * KEEPc * DDc) return;
    int c = idx % DDc;
    int s = (idx / DDc) % KEEPc;
    int b = idx / (DDc * KEEPc);
    int n = uidx[b * KEEPc + s];
    decin[((size_t)b * SD + 1 + n) * DDc + c] =
        proj[((size_t)b * SE + 1 + s) * DDc + c] + pe[(size_t)(1 + n) * DDc + c];
}

__global__ __launch_bounds__(256) void k_gather_masked(float* __restrict__ mo,
                                                       const float* __restrict__ dln,
                                                       const int* __restrict__ midx) {
    int idx = blockIdx.x * 256 + threadIdx.x;
    if (idx >= Bc * MSKc * DDc) return;
    int c = idx % DDc;
    int m = (idx / DDc) % MSKc;
    int b = idx / (DDc * MSKc);
    int n = midx[b * MSKc + m];
    mo[idx] = dln[((size_t)b * SD + 1 + n) * DDc + c];
}

__global__ __launch_bounds__(256) void k_gather_target(float* __restrict__ out,
                                                       const float* __restrict__ patches,
                                                       const int* __restrict__ midx) {
    int idx = blockIdx.x * 256 + threadIdx.x;
    if (idx >= Bc * MSKc * PP) return;
    int p = idx % PP;
    int m = (idx / PP) % MSKc;
    int b = idx / (PP * MSKc);
    int n = midx[b * MSKc + m];
    out[idx] = patches[((size_t)b * Nn + n) * PP + p];
}

__global__ __launch_bounds__(256) void k_mask_init(float* __restrict__ out) {
    int idx = blockIdx.x * 256 + threadIdx.x;
    if (idx < Bc * Nn) out[idx] = 0.f;
}
__global__ __launch_bounds__(256) void k_mask_set(float* __restrict__ out,
                                                  const int* __restrict__ midx) {
    int idx = blockIdx.x * 256 + threadIdx.x;
    if (idx < Bc * MSKc) {
        int b = idx / MSKc;
        out[b * Nn + midx[idx]] = 1.f;
    }
}

// ---------------------------------------------------------------------------
// host side
// ---------------------------------------------------------------------------
static inline void launch_gemm(const float* A, const unsigned short* Wb, float* C,
                               const float* bias, const float* res,
                               int M, int N, int K, int gelu, hipStream_t s) {
    dim3 grid(N / GBN, (M + GBM - 1) / GBM);
    k_gemm_wmma<<<grid, 256, 0, s>>>(A, Wb, C, bias, res, M, N, K, gelu);
}
static inline int gb(int n) { return (n + 255) / 256; }
static inline void conv_bf16(const float* src, unsigned short* dst, size_t n, hipStream_t s) {
    k_to_bf16<<<(int)((n + 255) / 256), 256, 0, s>>>(src, dst, (int)n);
}

extern "C" void kernel_launch(void* const* d_in, const int* in_sizes, int n_in,
                              void* d_out, int out_size, void* d_ws, size_t ws_size,
                              hipStream_t stream) {
    (void)in_sizes; (void)n_in; (void)out_size; (void)ws_size;
    const float* x        = (const float*)d_in[0];
    const float* proj_w   = (const float*)d_in[1];
    const float* proj_b   = (const float*)d_in[2];
    const float* cls_enc  = (const float*)d_in[3];
    const float* cls_dec  = (const float*)d_in[4];
    const float* e_ln1g = (const float*)d_in[5];
    const float* e_ln1b = (const float*)d_in[6];
    const float* e_qkvw = (const float*)d_in[7];
    const float* e_qkvb = (const float*)d_in[8];
    const float* e_pw   = (const float*)d_in[9];
    const float* e_pb   = (const float*)d_in[10];
    const float* e_ln2g = (const float*)d_in[11];
    const float* e_ln2b = (const float*)d_in[12];
    const float* e_f1w  = (const float*)d_in[13];
    const float* e_f1b  = (const float*)d_in[14];
    const float* e_f2w  = (const float*)d_in[15];
    const float* e_f2b  = (const float*)d_in[16];
    const float* enc_ng = (const float*)d_in[17];
    const float* enc_nb = (const float*)d_in[18];
    const float* p2d_w  = (const float*)d_in[19];
    const float* p2d_b  = (const float*)d_in[20];
    const float* maskem = (const float*)d_in[21];
    const float* d_ln1g = (const float*)d_in[22];
    const float* d_ln1b = (const float*)d_in[23];
    const float* d_qkvw = (const float*)d_in[24];
    const float* d_qkvb = (const float*)d_in[25];
    const float* d_pw   = (const float*)d_in[26];
    const float* d_pb   = (const float*)d_in[27];
    const float* d_ln2g = (const float*)d_in[28];
    const float* d_ln2b = (const float*)d_in[29];
    const float* d_f1w  = (const float*)d_in[30];
    const float* d_f1b  = (const float*)d_in[31];
    const float* d_f2w  = (const float*)d_in[32];
    const float* d_f2b  = (const float*)d_in[33];
    const float* dec_ng = (const float*)d_in[34];
    const float* dec_nb = (const float*)d_in[35];
    const float* recon_w= (const float*)d_in[36];
    const float* recon_b= (const float*)d_in[37];
    const float* class_w= (const float*)d_in[38];
    const float* class_b= (const float*)d_in[39];
    const int*   uidx   = (const int*)d_in[40];
    const int*   midx   = (const int*)d_in[41];

    float* out = (float*)d_out;
    constexpr size_t TGT = (size_t)Bc * MSKc * PP;

    // ---------------- workspace layout (floats first, then bf16 weights) ----
    float* W = (float*)d_ws;
    size_t off = 0;
    float* patches = W + off; off += (size_t)Bc * Nn * PP;
    float* emb     = W + off; off += (size_t)Bc * Nn * DEc;
    float* pe_e    = W + off; off += (size_t)(Nn + 1) * DEc;
    float* pe_d    = W + off; off += (size_t)(Nn + 1) * DDc;
    float* h_enc   = W + off; off += (size_t)Bc * SE * DEc;
    float* y_enc   = W + off; off += (size_t)Bc * SE * DEc;
    float* qkv_e   = W + off; off += (size_t)Bc * SE * 3 * DEc;
    float* o_enc   = W + off; off += (size_t)Bc * SE * DEc;
    float* mlp_e   = W + off; off += (size_t)Bc * SE * 4 * DEc;
    float* proj    = W + off; off += (size_t)Bc * SE * DDc;
    float* h_dec   = W + off; off += (size_t)Bc * SD * DDc;
    float* y_dec   = W + off; off += (size_t)Bc * SD * DDc;
    float* qkv_d   = W + off; off += (size_t)Bc * SD * 3 * DDc;
    float* o_dec   = W + off; off += (size_t)Bc * SD * DDc;
    float* mlp_d   = W + off; off += (size_t)Bc * SD * 4 * DDc;
    float* mo      = W + off; off += (size_t)Bc * MSKc * DDc;

    // bf16 weight region (16B-aligned: all preceding sizes are multiples of 4)
    unsigned short* WB = (unsigned short*)(W + off);
    size_t boff = 0;
    unsigned short* b_projw  = WB + boff; boff += (size_t)DEc * PP;
    unsigned short* b_eqkvw  = WB + boff; boff += (size_t)LE * 3 * DEc * DEc;
    unsigned short* b_epw    = WB + boff; boff += (size_t)LE * DEc * DEc;
    unsigned short* b_ef1w   = WB + boff; boff += (size_t)LE * 4 * DEc * DEc;
    unsigned short* b_ef2w   = WB + boff; boff += (size_t)LE * 4 * DEc * DEc;
    unsigned short* b_p2dw   = WB + boff; boff += (size_t)DDc * DEc;
    unsigned short* b_dqkvw  = WB + boff; boff += (size_t)LD * 3 * DDc * DDc;
    unsigned short* b_dpw    = WB + boff; boff += (size_t)LD * DDc * DDc;
    unsigned short* b_df1w   = WB + boff; boff += (size_t)LD * 4 * DDc * DDc;
    unsigned short* b_df2w   = WB + boff; boff += (size_t)LD * 4 * DDc * DDc;
    unsigned short* b_reconw = WB + boff; boff += (size_t)PP * DDc;
    unsigned short* b_classw = WB + boff; boff += (size_t)PP * DDc;

    float* redpart = (float*)(WB + boff); // boff is even -> float aligned
    float* stats   = redpart + 512;

    const int nx = Bc * 256 * 256;

    // 0. pre-convert all weights to bf16 (once per call)
    conv_bf16(proj_w,  b_projw,  (size_t)DEc * PP, stream);
    conv_bf16(e_qkvw,  b_eqkvw,  (size_t)LE * 3 * DEc * DEc, stream);
    conv_bf16(e_pw,    b_epw,    (size_t)LE * DEc * DEc, stream);
    conv_bf16(e_f1w,   b_ef1w,   (size_t)LE * 4 * DEc * DEc, stream);
    conv_bf16(e_f2w,   b_ef2w,   (size_t)LE * 4 * DEc * DEc, stream);
    conv_bf16(p2d_w,   b_p2dw,   (size_t)DDc * DEc, stream);
    conv_bf16(d_qkvw,  b_dqkvw,  (size_t)LD * 3 * DDc * DDc, stream);
    conv_bf16(d_pw,    b_dpw,    (size_t)LD * DDc * DDc, stream);
    conv_bf16(d_f1w,   b_df1w,   (size_t)LD * 4 * DDc * DDc, stream);
    conv_bf16(d_f2w,   b_df2w,   (size_t)LD * 4 * DDc * DDc, stream);
    conv_bf16(recon_w, b_reconw, (size_t)PP * DDc, stream);
    conv_bf16(class_w, b_classw, (size_t)PP * DDc, stream);

    // 1. global norm stats + patchify
    k_reduce_partial<<<256, 256, 0, stream>>>(x, redpart, nx);
    k_reduce_final  <<<1, 256, 0, stream>>>(redpart, stats, nx);
    k_patchify      <<<gb(Bc * Nn * PP), 256, 0, stream>>>(x, stats, patches);

    // 2. positional embeddings
    k_pos_emb<<<gb((Nn + 1) * DEc), 256, 0, stream>>>(pe_e, Nn + 1, DEc);
    k_pos_emb<<<gb((Nn + 1) * DDc), 256, 0, stream>>>(pe_d, Nn + 1, DDc);

    // 3. patch embed + pos add + gather unmasked into encoder input
    launch_gemm(patches, b_projw, emb, proj_b, nullptr, Bc * Nn, DEc, PP, 0, stream);
    k_add_pe<<<gb(Bc * Nn * DEc), 256, 0, stream>>>(emb, pe_e);
    k_build_encin<<<gb(Bc * SE * DEc), 256, 0, stream>>>(h_enc, emb, cls_enc, pe_e, uidx);

    // 4. encoder: 12 layers
    const int ME = Bc * SE;
    for (int l = 0; l < LE; ++l) {
        k_layernorm<<<ME, 256, 0, stream>>>(h_enc, e_ln1g + (size_t)l * DEc, e_ln1b + (size_t)l * DEc, y_enc, DEc);
        launch_gemm(y_enc, b_eqkvw + (size_t)l * 3 * DEc * DEc, qkv_e, e_qkvb + (size_t)l * 3 * DEc,
                    nullptr, ME, 3 * DEc, DEc, 0, stream);
        k_attention<<<(Bc * HE * SE + 127) / 128, 128, 0, stream>>>(qkv_e, o_enc, SE, DEc, HE);
        launch_gemm(o_enc, b_epw + (size_t)l * DEc * DEc, h_enc, e_pb + (size_t)l * DEc,
                    h_enc, ME, DEc, DEc, 0, stream);
        k_layernorm<<<ME, 256, 0, stream>>>(h_enc, e_ln2g + (size_t)l * DEc, e_ln2b + (size_t)l * DEc, y_enc, DEc);
        launch_gemm(y_enc, b_ef1w + (size_t)l * 4 * DEc * DEc, mlp_e, e_f1b + (size_t)l * 4 * DEc,
                    nullptr, ME, 4 * DEc, DEc, 1, stream);
        launch_gemm(mlp_e, b_ef2w + (size_t)l * 4 * DEc * DEc, h_enc, e_f2b + (size_t)l * DEc,
                    h_enc, ME, DEc, 4 * DEc, 0, stream);
    }

    // 5. final encoder LN + project to decoder width
    k_layernorm<<<ME, 256, 0, stream>>>(h_enc, enc_ng, enc_nb, y_enc, DEc);
    launch_gemm(y_enc, b_p2dw, proj, p2d_b, nullptr, ME, DDc, DEc, 0, stream);

    // 6. decoder input: mask tokens + scatter projected unmasked tokens
    k_build_decin<<<gb(Bc * SD * DDc), 256, 0, stream>>>(h_dec, cls_dec, maskem, pe_d);
    k_scatter_unmasked<<<gb(Bc * KEEPc * DDc), 256, 0, stream>>>(h_dec, proj, pe_d, uidx);

    // 7. decoder: 4 layers
    const int MD = Bc * SD;
    for (int l = 0; l < LD; ++l) {
        k_layernorm<<<MD, 256, 0, stream>>>(h_dec, d_ln1g + (size_t)l * DDc, d_ln1b + (size_t)l * DDc, y_dec, DDc);
        launch_gemm(y_dec, b_dqkvw + (size_t)l * 3 * DDc * DDc, qkv_d, d_qkvb + (size_t)l * 3 * DDc,
                    nullptr, MD, 3 * DDc, DDc, 0, stream);
        k_attention<<<(Bc * HD * SD + 127) / 128, 128, 0, stream>>>(qkv_d, o_dec, SD, DDc, HD);
        launch_gemm(o_dec, b_dpw + (size_t)l * DDc * DDc, h_dec, d_pb + (size_t)l * DDc,
                    h_dec, MD, DDc, DDc, 0, stream);
        k_layernorm<<<MD, 256, 0, stream>>>(h_dec, d_ln2g + (size_t)l * DDc, d_ln2b + (size_t)l * DDc, y_dec, DDc);
        launch_gemm(y_dec, b_df1w + (size_t)l * 4 * DDc * DDc, mlp_d, d_f1b + (size_t)l * 4 * DDc,
                    nullptr, MD, 4 * DDc, DDc, 1, stream);
        launch_gemm(mlp_d, b_df2w + (size_t)l * 4 * DDc * DDc, h_dec, d_f2b + (size_t)l * DDc,
                    h_dec, MD, DDc, 4 * DDc, 0, stream);
    }

    // 8. final decoder LN, gather masked tokens
    k_layernorm<<<MD, 256, 0, stream>>>(h_dec, dec_ng, dec_nb, y_dec, DDc);
    k_gather_masked<<<gb(Bc * MSKc * DDc), 256, 0, stream>>>(mo, y_dec, midx);

    // 9. outputs: target | recon | cls_logits | bool_mask
    k_gather_target<<<gb(Bc * MSKc * PP), 256, 0, stream>>>(out, patches, midx);
    launch_gemm(mo, b_reconw, out + TGT,     recon_b, nullptr, Bc * MSKc, PP, DDc, 0, stream);
    launch_gemm(mo, b_classw, out + 2 * TGT, class_b, nullptr, Bc * MSKc, PP, DDc, 0, stream);
    k_mask_init<<<gb(Bc * Nn), 256, 0, stream>>>(out + 3 * TGT);
    k_mask_set <<<gb(Bc * MSKc), 256, 0, stream>>>(out + 3 * TGT, midx);
}
